// Network_13314398617919
// MI455X (gfx1250) — compile-verified, matrix operand order checked by reading, not run on hardware
//
#include <hip/hip_runtime.h>

#define Bx 32
#define Sx 2048
#define Hx 512
#define Ex 128
#define Dx 15

typedef __attribute__((ext_vector_type(16))) _Float16 v16h;
typedef __attribute__((ext_vector_type(8)))  float    v8f;

__device__ __forceinline__ float sigf(float x) {
    // 1 / (1 + exp(-x)) : v_exp_f32 + v_rcp_f32, stable at +-inf
    return __builtin_amdgcn_rcpf(1.0f + __expf(-x));
}
__device__ __forceinline__ float tanhfast(float x) {
    // tanh(x) = 2*sigmoid(2x) - 1, stable at +-inf
    return fmaf(2.0f, sigf(2.0f * x), -1.0f);
}

// ---------------------------------------------------------------------------
// Kernel 0: collapse the (no-nonlinearity) MLP into M (15x15) and c (15).
//   M[k][n] = sum_e W1[e,k] * W2[n,e]   stored f16, padded to 16x32, as
//   column-per-lane layout source (Mt[n*32+k]) for the WMMA B operand.
// ---------------------------------------------------------------------------
__global__ void prep_kernel(const float* __restrict__ W1, const float* __restrict__ b1,
                            const float* __restrict__ W2, const float* __restrict__ b2,
                            _Float16* __restrict__ Mt, float* __restrict__ cvec) {
    int t = threadIdx.x;
    if (t < 512) {
        int n = t >> 5, kk = t & 31;
        float v = 0.f;
        if (n < Dx && kk < Dx) {
            for (int e = 0; e < Ex; ++e) v = fmaf(W1[e * Dx + kk], W2[n * Ex + e], v);
        }
        Mt[n * 32 + kk] = (_Float16)v;
    }
    if (t < 16) {
        float cv = 0.f;
        if (t < Dx) {
            for (int e = 0; e < Ex; ++e) cv = fmaf(b1[e], W2[t * Ex + e], cv);
            cv += b2[t];
        }
        cvec[t] = cv;
    }
}

// ---------------------------------------------------------------------------
// Kernel 1: forecasts = x @ M + c via v_wmma_f32_16x16x32_f16.
// One wave per 16-row tile of x (65536 rows -> 4096 tiles, 8 waves/block).
// ---------------------------------------------------------------------------
__global__ void __launch_bounds__(256) forecast_kernel(const float* __restrict__ x,
                                                       const _Float16* __restrict__ Mt,
                                                       const float* __restrict__ cvec,
                                                       float* __restrict__ fout) {
    const int lane = threadIdx.x & 31;
    const int wave = threadIdx.x >> 5;
    const int tile = blockIdx.x * 8 + wave;
    const int r0   = tile * 16;
    const int m    = lane & 15;   // A: row; C/D & B: column
    const int hi   = lane >> 4;   // K-half for A/B, M-half for C/D

    v16h a, b;
    const float*    xr = x + (size_t)(r0 + m) * Dx;
    const _Float16* bm = Mt + m * 32;
#pragma unroll
    for (int v = 0; v < 8; ++v) {
        // 16-bit A layout: VGPR v holds K = k0, k0+1 (ISA 7.12.2)
        int k0 = ((v < 4) ? 0 : 16) + hi * 8 + (v & 3) * 2;
        a[2 * v]     = (k0     < Dx) ? (_Float16)xr[k0]     : (_Float16)0.f;
        a[2 * v + 1] = (k0 + 1 < Dx) ? (_Float16)xr[k0 + 1] : (_Float16)0.f;
        b[2 * v]     = bm[k0];
        b[2 * v + 1] = bm[k0 + 1];
    }
    v8f c = {};
    c = __builtin_amdgcn_wmma_f32_16x16x32_f16(false, a, false, b, (short)0, c, false, false);

    if (m < Dx) {
        float cb = cvec[m];
#pragma unroll
        for (int r = 0; r < 8; ++r) {
            int row = r0 + hi * 8 + r;             // C/D layout: VGPR r -> M = r + 8*hi
            fout[(size_t)row * Dx + m] = c[r] + cb;
        }
    }
}

// ---------------------------------------------------------------------------
// Kernel 2: the fully sequential scan. 67,584 dependent LSTM cells.
// 512 threads (16 wave32s): thread k owns gate rows (k, 512+k, 1024+k, 1536+k)
// = (i,f,g,o) for H-index k, so c[k] never leaves a register.
// Software-pipelined: x-dots (+bias) for cell N+1 computed before cell N's
// barrier, so the post-barrier chain is h -> 1 FMA -> activations -> reduce.
// One __syncthreads per cell (parity double-buffered LDS partials).
// ---------------------------------------------------------------------------
__global__ void __launch_bounds__(512) lstm_kernel(const float* __restrict__ x,
                                                   const float* __restrict__ forec,
                                                   const float* __restrict__ W_ih,
                                                   const float* __restrict__ W_hh,
                                                   const float* __restrict__ b_ih,
                                                   const float* __restrict__ b_hh,
                                                   const float* __restrict__ W_hr,
                                                   float* __restrict__ prog,
                                                   float* __restrict__ fore) {
    const int k    = threadIdx.x;
    const int lane = k & 31;
    const int wid  = k >> 5;

    float wi[Dx], wf[Dx], wg[Dx], wo[Dx];
#pragma unroll
    for (int d = 0; d < Dx; ++d) {
        wi[d] = W_ih[(0 * Hx + k) * Dx + d];
        wf[d] = W_ih[(1 * Hx + k) * Dx + d];
        wg[d] = W_ih[(2 * Hx + k) * Dx + d];
        wo[d] = W_ih[(3 * Hx + k) * Dx + d];
    }
    const float bi_ = b_ih[0 * Hx + k] + b_hh[0 * Hx + k];
    const float bf_ = b_ih[1 * Hx + k] + b_hh[1 * Hx + k];
    const float bg_ = b_ih[2 * Hx + k] + b_hh[2 * Hx + k];
    const float bo_ = b_ih[3 * Hx + k] + b_hh[3 * Hx + k];
    const float whi = W_hh[0 * Hx + k], whf = W_hh[1 * Hx + k];
    const float whg = W_hh[2 * Hx + k], who = W_hh[3 * Hx + k];
    const float whr = W_hr[k];

    __shared__ float partial[2][16];

    // x-dots + bias for an input row (h-independent part of a cell)
    auto dot4 = [&](const float* __restrict__ xr,
                    float& a, float& b, float& c, float& d) {
        float ra = bi_, rb = bf_, rc = bg_, rd = bo_;
#pragma unroll
        for (int t = 0; t < Dx; ++t) {
            float xv = xr[t];   // uniform address -> scalar load
            ra = fmaf(wi[t], xv, ra);
            rb = fmaf(wf[t], xv, rb);
            rc = fmaf(wg[t], xv, rc);
            rd = fmaf(wo[t], xv, rd);
        }
        a = ra; b = rb; c = rc; d = rd;
    };
    // depth-4 tree sum of the 16 wave partials (LDS reads issue in parallel)
    auto sum16 = [&](const float* __restrict__ q) -> float {
        float s0 = q[0] + q[1],   s1 = q[2] + q[3];
        float s2 = q[4] + q[5],   s3 = q[6] + q[7];
        float s4 = q[8] + q[9],   s5 = q[10] + q[11];
        float s6 = q[12] + q[13], s7 = q[14] + q[15];
        float a = s0 + s1, b = s2 + s3, c = s4 + s5, d = s6 + s7;
        return (a + b) + (c + d);
    };

    float h = 0.f, ck = 0.f;
    int p = 0;

    // prologue: dots for cell (s=0, b=0)
    float pi, pf, pg, po;
    dot4(x, pi, pf, pg, po);

    for (int s = 0; s < Sx; ++s) {
        for (int b = 0; b < Bx; ++b) {
            // ---- recurrent part of cell (s, b): minimal post-barrier chain
            float gi = fmaf(whi, h, pi);
            float gf = fmaf(whf, h, pf);
            float gg = fmaf(whg, h, pg);
            float go = fmaf(who, h, po);
            float ai = sigf(gi), af = sigf(gf), ao = sigf(go);
            float tg = tanhfast(gg);
            ck = fmaf(af, ck, ai * tg);
            float val = ao * tanhfast(ck) * whr;
#pragma unroll
            for (int off = 16; off > 0; off >>= 1) val += __shfl_xor(val, off, 32);
            if (lane == 0) partial[p][wid] = val;

            // ---- pipeline: dots for the NEXT cell (b+1, or forecast row s)
            if (k == 0 && b + 2 < Bx)
                __builtin_prefetch(x + ((size_t)(b + 2) * Sx + s) * Dx, 0, 1);
            const float* nr = (b + 1 < Bx) ? x + ((size_t)(b + 1) * Sx + s) * Dx
                                           : forec + (size_t)s * Dx;
            float ni, nf, ng, no;
            dot4(nr, ni, nf, ng, no);

            __syncthreads();
            h = sum16(partial[p]);
            p ^= 1;
            if (k == 0) prog[(size_t)b * Sx + s] = h;
            pi = ni; pf = nf; pg = ng; po = no;
        }
        // ---- forecast cell: uses current (h, ck) but does NOT update them
        {
            float gi = fmaf(whi, h, pi);
            float gf = fmaf(whf, h, pf);
            float gg = fmaf(whg, h, pg);
            float go = fmaf(who, h, po);
            float ai = sigf(gi), af = sigf(gf), ao = sigf(go);
            float tg = tanhfast(gg);
            float c2 = fmaf(af, ck, ai * tg);          // local only
            float val = ao * tanhfast(c2) * whr;
#pragma unroll
            for (int off = 16; off > 0; off >>= 1) val += __shfl_xor(val, off, 32);
            if (lane == 0) partial[p][wid] = val;

            // pipeline: dots for (s+1, b=0)
            const float* nr = x + (size_t)((s + 1 < Sx) ? (s + 1) : s) * Dx;
            float ni, nf, ng, no;
            dot4(nr, ni, nf, ng, no);

            __syncthreads();
            float hf = sum16(partial[p]);
            p ^= 1;
            if (k < Bx) fore[(size_t)k * Sx + s] = hf;  // forecasted[b][s] = fprog[s]
            pi = ni; pf = nf; pg = ng; po = no;
        }
    }
}

extern "C" void kernel_launch(void* const* d_in, const int* in_sizes, int n_in,
                              void* d_out, int out_size, void* d_ws, size_t ws_size,
                              hipStream_t stream) {
    (void)in_sizes; (void)n_in; (void)out_size; (void)ws_size;
    const float* x    = (const float*)d_in[0];
    const float* W1   = (const float*)d_in[1];
    const float* b1   = (const float*)d_in[2];
    const float* W2   = (const float*)d_in[3];
    const float* b2   = (const float*)d_in[4];
    const float* W_ih = (const float*)d_in[5];
    const float* W_hh = (const float*)d_in[6];
    const float* b_ih = (const float*)d_in[7];
    const float* b_hh = (const float*)d_in[8];
    const float* W_hr = (const float*)d_in[9];

    float* out   = (float*)d_out;
    float* prog  = out;                        // progress   (B, S)
    float* fore  = out + (size_t)Bx * Sx;      // forecasted (B, S)
    float* forec = out + (size_t)2 * Bx * Sx;  // forecasts  (1, B*S, D)

    _Float16* Mt  = (_Float16*)d_ws;                     // 16x32 f16 = 1 KB
    float*    cvc = (float*)((char*)d_ws + 1024);        // 16 f32

    prep_kernel<<<1, 512, 0, stream>>>(W1, b1, W2, b2, Mt, cvc);

    // 65536 rows / 16 per wave / 8 waves per block = 512 blocks
    forecast_kernel<<<512, 256, 0, stream>>>(x, Mt, cvc, forec);

    lstm_kernel<<<1, 512, 0, stream>>>(x, forec, W_ih, W_hh, b_ih, b_hh, W_hr,
                                       prog, fore);
}
